// WaypointTransformer_49718541418550
// MI455X (gfx1250) — matrix operand order repaired and checked by reading, not compiled, not run on hardware
//
#include <hip/hip_runtime.h>
#include <hip/hip_bf16.h>

// ---------------- constants ----------------
#define BLOCK   256
#define NWAVES  8
#define NSEQ    8          // sequences per block
#define TT      10         // tokens per sequence
#define DD      64         // model dim
#define FF      256        // ffn hidden
#define NLAYER  3
#define MROWS   (NSEQ*TT)  // 80 token rows per block
#define MT      5          // MROWS / 16
#define SHH     72         // row stride (elems) for h (f32) and hb (bf16); 144B rows, 16B aligned
#define SZZ     272        // row stride (elems) for zb (bf16); 544B rows, 16B aligned
#define SW2     264        // row stride for w2^T staging (K=256); 528B rows, 16B aligned

typedef __bf16 bf16_t;
typedef __attribute__((ext_vector_type(16))) __bf16 v16bf;
typedef __attribute__((ext_vector_type(8)))  __bf16 v8bf;
typedef __attribute__((ext_vector_type(8)))  float  v8f;

union V16U { v16bf v; v8bf h[2]; };

// ---- A fragment: 16x32 bf16, row-major LDS (stride sa elems) ----
// ISA layout: lanes 0-15 row M=lane, VGPR0-3 = K 0..7, VGPR4-7 = K 16..23;
//             lanes 16-31 same rows, K 8..15 / 24..31.
__device__ __forceinline__ v16bf load_a_frag(const bf16_t* A, int sa, int rt, int kt, int lane) {
    int m = lane & 15, half = lane >> 4;
    const bf16_t* p = A + (rt * 16 + m) * sa + kt * 32 + half * 8;
    V16U u;
    u.h[0] = *(const v8bf*)(p);        // K +0..7   (of this half's base)
    u.h[1] = *(const v8bf*)(p + 16);   // K +16..23
    return u.v;
}

// ---- B fragment: 32x16 bf16. Weights staged transposed (N x K row-major, stride sb). ----
// ISA layout: lanes 0-15 col N=lane hold K 0..7 / 16..23; lanes 16-31 hold K 8..15 / 24..31.
__device__ __forceinline__ v16bf load_b_frag(const bf16_t* Bm, int sb, int ct, int kt, int lane) {
    int n = lane & 15, half = lane >> 4;
    const bf16_t* p = Bm + (ct * 16 + n) * sb + kt * 32 + half * 8;
    V16U u;
    u.h[0] = *(const v8bf*)(p);        // K +0..7   (of this half's base)
    u.h[1] = *(const v8bf*)(p + 16);   // K +16..23
    return u.v;
}

// Two independent output tiles per call: their WMMAs interleave, hiding the
// 16-bit WMMA RAW hazard slots and doubling LDS loads in flight.
__device__ __forceinline__ void tile_mm2(const bf16_t* A, int sa, const bf16_t* Bm, int sb,
                                         int rt1, int ct1, int rt2, int ct2, bool has2,
                                         int ktiles, int lane, v8f& c1, v8f& c2) {
    for (int kt = 0; kt < ktiles; ++kt) {
        v16bf a1 = load_a_frag(A, sa, rt1, kt, lane);
        v16bf b1 = load_b_frag(Bm, sb, ct1, kt, lane);
        v16bf a2 = {}, b2 = {};
        if (has2) {
            a2 = load_a_frag(A, sa, rt2, kt, lane);
            b2 = load_b_frag(Bm, sb, ct2, kt, lane);
        }
        c1 = __builtin_amdgcn_wmma_f32_16x16x32_bf16(false, a1, false, b1,
                                                     (short)0, c1, false, false);
        if (has2)
            c2 = __builtin_amdgcn_wmma_f32_16x16x32_bf16(false, a2, false, b2,
                                                         (short)0, c2, false, false);
    }
}

// C/D layout: VGPR i -> row = i + 8*(lane>=16), col = lane&15.
__device__ __forceinline__ void store_tile_bf16(v8f c, bf16_t* dst, int sd, int nt, int tile,
                                                const float* bias, int do_gelu, int lane) {
    int rt = tile / nt, ct = tile - rt * nt;
    int col   = ct * 16 + (lane & 15);
    int rbase = rt * 16 + 8 * (lane >> 4);
    if (bias) {
        float bv = bias[col];
#pragma unroll
        for (int i = 0; i < 8; ++i) {
            float v = c[i] + bv;
            if (do_gelu) v = 0.5f * v * (1.0f + erff(v * 0.70710678118f));  // exact gelu
            dst[(rbase + i) * sd + col] = (bf16_t)v;
        }
    } else {
#pragma unroll
        for (int i = 0; i < 8; ++i)
            dst[(rbase + i) * sd + col] = (bf16_t)c[i];
    }
}

__device__ __forceinline__ void accum_tile_f32(v8f c, float* dst, int sd, int nt, int tile,
                                               const float* bias, int lane) {
    int rt = tile / nt, ct = tile - rt * nt;
    int col   = ct * 16 + (lane & 15);
    int rbase = rt * 16 + 8 * (lane >> 4);
    float bv = bias ? bias[col] : 0.0f;
#pragma unroll
    for (int i = 0; i < 8; ++i)
        dst[(rbase + i) * sd + col] += c[i] + bv;   // tile owned by one wave: race-free
}

__device__ void gemm_store_bf16(const bf16_t* A, int sa, const bf16_t* Bm, int sb,
                                bf16_t* dst, int sd, int mt, int nt, int ktiles,
                                const float* bias, int do_gelu, int wave, int lane) {
    int total = mt * nt;
    for (int tile = wave; tile < total; tile += 2 * NWAVES) {
        int t2 = tile + NWAVES;
        bool has2 = t2 < total;
        int rt1 = tile / nt, ct1 = tile - rt1 * nt;
        int rt2 = 0, ct2 = 0;
        if (has2) { rt2 = t2 / nt; ct2 = t2 - rt2 * nt; }
        v8f c1 = {}, c2 = {};
        tile_mm2(A, sa, Bm, sb, rt1, ct1, rt2, ct2, has2, ktiles, lane, c1, c2);
        store_tile_bf16(c1, dst, sd, nt, tile, bias, do_gelu, lane);
        if (has2) store_tile_bf16(c2, dst, sd, nt, t2, bias, do_gelu, lane);
    }
}

__device__ void gemm_accum_f32(const bf16_t* A, int sa, const bf16_t* Bm, int sb,
                               float* dst, int sd, int mt, int nt, int ktiles,
                               const float* bias, int wave, int lane) {
    int total = mt * nt;
    for (int tile = wave; tile < total; tile += 2 * NWAVES) {
        int t2 = tile + NWAVES;
        bool has2 = t2 < total;
        int rt1 = tile / nt, ct1 = tile - rt1 * nt;
        int rt2 = 0, ct2 = 0;
        if (has2) { rt2 = t2 / nt; ct2 = t2 - rt2 * nt; }
        v8f c1 = {}, c2 = {};
        tile_mm2(A, sa, Bm, sb, rt1, ct1, rt2, ct2, has2, ktiles, lane, c1, c2);
        accum_tile_f32(c1, dst, sd, nt, tile, bias, lane);
        if (has2) accum_tile_f32(c2, dst, sd, nt, t2, bias, lane);
    }
}

// stage W[K][N] (f32 global, row-major) -> W^T[N][K] bf16 in LDS with stride sw
__device__ __forceinline__ void stage_wt(const float* W, int K, int N, bf16_t* wb, int sw, int tid) {
    for (int idx = tid; idx < K * N; idx += BLOCK) {
        int k = idx / N, n = idx - k * N;
        wb[n * sw + k] = (bf16_t)W[idx];
    }
}

__device__ __forceinline__ void layernorm_rows(const float* hsrc, bf16_t* dst,
                                               const float* g, const float* b, int tid) {
    for (int r = tid; r < MROWS; r += BLOCK) {
        const float* row = hsrc + r * SHH;
        float m = 0.f;
        for (int c = 0; c < DD; ++c) m += row[c];
        m *= (1.0f / DD);
        float v = 0.f;
        for (int c = 0; c < DD; ++c) { float d = row[c] - m; v += d * d; }
        v *= (1.0f / DD);
        float inv = rsqrtf(v + 1e-5f);
        bf16_t* drow = dst + r * SHH;
        for (int c = 0; c < DD; ++c)
            drow[c] = (bf16_t)((row[c] - m) * inv * g[c] + b[c]);
    }
}

// causal multi-head attention on qkv in zb ([80][SZZ] bf16: q|k|v at cols 0/64/128),
// writes y (bf16) into hb. Two head-instances per wave pass: one per 16-lane half.
__device__ void attention_heads(const bf16_t* qkvb, bf16_t* hb, int wave, int lane) {
    const int sub = lane >> 4;          // which half-wave -> which instance
    const int t   = lane & 15;          // query token within instance
#pragma unroll 1
    for (int ii = 0; ii < 2; ++ii) {    // 2 passes x 2 halves = 4 instances per wave
        int inst = wave * 4 + ii * 2 + sub;      // 0..31 = (seq, head)
        int s = inst >> 2, hd = inst & 3;
        if (t < TT) {
            const int rowq = (s * TT + t) * SZZ + hd * 16;
            float q[16];
#pragma unroll
            for (int d = 0; d < 16; ++d) q[d] = (float)qkvb[rowq + d];

            // scores (rows read unconditionally -> contiguous b64/b128 merges; mask after)
            float sc[TT]; float mx = -1e30f;
#pragma unroll
            for (int k = 0; k < TT; ++k) {
                const bf16_t* kr = qkvb + (s * TT + k) * SZZ + DD + hd * 16;
                float acc = 0.f;
#pragma unroll
                for (int d = 0; d < 16; ++d) acc += q[d] * (float)kr[d];
                float a = (k <= t) ? acc * 0.25f : -1e9f;   // causal mask, dh^-0.5 = 1/4
                sc[k] = a; mx = fmaxf(mx, a);
            }
            float sum = 0.f;
#pragma unroll
            for (int k = 0; k < TT; ++k) { float e = __expf(sc[k] - mx); sc[k] = e; sum += e; }
            float inv = 1.0f / sum;

            // y = P @ V : k-outer, register accumulator, contiguous V-row reads
            float y[16];
#pragma unroll
            for (int d = 0; d < 16; ++d) y[d] = 0.f;
#pragma unroll
            for (int k = 0; k < TT; ++k) {
                const bf16_t* vr = qkvb + (s * TT + k) * SZZ + 2 * DD + hd * 16;
                float w = sc[k];                 // ~0 for masked k
#pragma unroll
                for (int d = 0; d < 16; ++d) y[d] += w * (float)vr[d];
            }
            bf16_t* yr = hb + (s * TT + t) * SHH + hd * 16;
#pragma unroll
            for (int d = 0; d < 16; ++d) yr[d] = (bf16_t)(y[d] * inv);
        }
    }
}

extern "C" __global__ __launch_bounds__(BLOCK)
void wp_transformer(const float* __restrict__ x,
                    const float* __restrict__ ew,   const float* __restrict__ eb,
                    const float* __restrict__ ln1g, const float* __restrict__ ln1b,
                    const float* __restrict__ qkvw, const float* __restrict__ projw,
                    const float* __restrict__ ln2g, const float* __restrict__ ln2b,
                    const float* __restrict__ w1,   const float* __restrict__ b1,
                    const float* __restrict__ w2,   const float* __restrict__ b2,
                    const float* __restrict__ lnfg, const float* __restrict__ lnfb,
                    const float* __restrict__ headw,const float* __restrict__ headb,
                    float* __restrict__ out) {
    extern __shared__ __attribute__((aligned(16))) char smem[];
    float*  hres = (float*) (smem);            // [80][72] f32 residual      23040 B
    bf16_t* hb   = (bf16_t*)(smem + 23040);    // [80][72] bf16 GEMM input   11520 B
    bf16_t* zb   = (bf16_t*)(smem + 34560);    // [80][272] bf16 qkv/ffn     43520 B
    bf16_t* wb   = (bf16_t*)(smem + 78080);    // weight staging (max 18432) 36864 B

    const int tid = threadIdx.x, lane = tid & 31, wave = tid >> 5;
    const int blk = blockIdx.x;

    // ---- embed: h = x @ embed_w + embed_b ----
    for (int r = tid; r < MROWS; r += BLOCK) {
        int s = r / TT, t = r - s * TT;
        const float* xp = x + ((size_t)(blk * NSEQ + s) * TT + t) * 4;
        float x0 = xp[0], x1 = xp[1], x2 = xp[2], x3 = xp[3];
        float* hr = hres + r * SHH;
        for (int c = 0; c < DD; ++c)
            hr[c] = x0 * ew[c] + x1 * ew[DD + c] + x2 * ew[2 * DD + c] + x3 * ew[3 * DD + c] + eb[c];
    }
    __syncthreads();

    for (int li = 0; li < NLAYER; ++li) {
        const float* qw  = qkvw + (size_t)li * DD * 3 * DD;
        const float* pw  = projw + (size_t)li * DD * DD;
        const float* w1l = w1 + (size_t)li * DD * FF;
        const float* w2l = w2 + (size_t)li * FF * DD;

        // ln1 -> hb ; stage qkv^T [192][72]
        layernorm_rows(hres, hb, ln1g + li * DD, ln1b + li * DD, tid);
        stage_wt(qw, DD, 3 * DD, wb, SHH, tid);
        // warm cache for FFN weights (global_prefetch_b8)
        {
            size_t off = (size_t)tid * 64;
            if (off < (size_t)DD * FF) {
                __builtin_prefetch(w1l + off, 0, 2);
                __builtin_prefetch(w2l + off, 0, 2);
            }
        }
        __syncthreads();

        // qkv = ln1(h) @ Wqkv  -> zb [80][192]
        gemm_store_bf16(hb, SHH, wb, SHH, zb, SZZ, MT, 12, 2, nullptr, 0, wave, lane);
        __syncthreads();

        // attention -> hb ; stage proj^T [64][72]
        attention_heads(zb, hb, wave, lane);
        stage_wt(pw, DD, DD, wb, SHH, tid);
        __syncthreads();

        // h += y @ Wproj
        gemm_accum_f32(hb, SHH, wb, SHH, hres, SHH, MT, 4, 2, nullptr, wave, lane);
        __syncthreads();

        // ln2 -> hb ; stage w1^T [256][72]
        layernorm_rows(hres, hb, ln2g + li * DD, ln2b + li * DD, tid);
        stage_wt(w1l, DD, FF, wb, SHH, tid);
        __syncthreads();

        // z = gelu(ln2(h) @ W1 + b1) -> zb [80][256]
        gemm_store_bf16(hb, SHH, wb, SHH, zb, SZZ, MT, 16, 2, b1 + li * FF, 1, wave, lane);
        __syncthreads();

        // stage w2^T [64][264]
        stage_wt(w2l, FF, DD, wb, SW2, tid);
        __syncthreads();

        // h += z @ W2 + b2   (K = 256 -> 8 k-tiles)
        gemm_accum_f32(zb, SZZ, wb, SW2, hres, SHH, MT, 4, 8, b2 + li * DD, wave, lane);
        __syncthreads();
    }

    // final LN -> hb, then head on last token of each sequence
    layernorm_rows(hres, hb, lnfg, lnfb, tid);
    __syncthreads();

    for (int i = tid; i < NSEQ * 10; i += BLOCK) {   // 2*SEQ_OUT = 10 outputs/seq
        int s = i / 10, o = i - s * 10;
        const bf16_t* row = hb + (s * TT + (TT - 1)) * SHH;
        float acc = headb[o];
        for (int d = 0; d < DD; ++d) acc += (float)row[d] * headw[d * 10 + o];
        out[((size_t)blk * NSEQ + s) * 10 + o] = acc;
    }
}

extern "C" void kernel_launch(void* const* d_in, const int* in_sizes, int n_in,
                              void* d_out, int out_size, void* d_ws, size_t ws_size,
                              hipStream_t stream) {
    (void)in_sizes; (void)n_in; (void)d_ws; (void)ws_size; (void)out_size;
    const float* x     = (const float*)d_in[0];
    const float* ew    = (const float*)d_in[1];
    const float* eb    = (const float*)d_in[2];
    const float* ln1g  = (const float*)d_in[3];
    const float* ln1b  = (const float*)d_in[4];
    const float* qkvw  = (const float*)d_in[5];
    const float* projw = (const float*)d_in[6];
    const float* ln2g  = (const float*)d_in[7];
    const float* ln2b  = (const float*)d_in[8];
    const float* w1    = (const float*)d_in[9];
    const float* b1    = (const float*)d_in[10];
    const float* w2    = (const float*)d_in[11];
    const float* b2    = (const float*)d_in[12];
    const float* lnfg  = (const float*)d_in[13];
    const float* lnfb  = (const float*)d_in[14];
    const float* headw = (const float*)d_in[15];
    const float* headb = (const float*)d_in[16];
    float* out = (float*)d_out;

    const int nblocks = 16384 / NSEQ;              // 2048
    const size_t smem = 23040 + 11520 + 43520 + 36864;  // 114944 B dynamic LDS
    wp_transformer<<<nblocks, BLOCK, smem, stream>>>(
        x, ew, eb, ln1g, ln1b, qkvw, projw, ln2g, ln2b,
        w1, b1, w2, b2, lnfg, lnfb, headw, headb, out);
}
